// FrameTokenizer_55903294325289
// MI455X (gfx1250) — compile-verified, matrix operand order checked by reading, not run on hardware
//
#include <hip/hip_runtime.h>

// ---------------------------------------------------------------------------
// FrameTokenizer for MI455X (gfx1250):
//   pool -> bf16 GEMM (TDM -> LDS double-buffer -> WMMA) -> LayerNorm
// ---------------------------------------------------------------------------

#define Bq      8
#define Tq      3000
#define Nq      512
#define ENC     1024
#define TOK     4096
#define MTOT    (Bq * Nq)        // 4096 rows (tokens)
#define FRATE   50.0f
#define LN_EPS  1e-5f

#define KSTEP   64               // K-chunk staged in LDS per buffer
#define NKB     (ENC / KSTEP)    // 16 K-chunks
#define BM      64               // block tile M
#define BN      128              // block tile N

typedef __attribute__((ext_vector_type(16))) __bf16 v16bf;
typedef __attribute__((ext_vector_type(8)))  __bf16 v8bf;
typedef __attribute__((ext_vector_type(8)))  float  v8f;
typedef __attribute__((ext_vector_type(4)))  unsigned int v4u;
typedef __attribute__((ext_vector_type(8)))  int v8i;
typedef __attribute__((ext_vector_type(4)))  int v4i;

#if defined(__has_builtin)
#  if __has_builtin(__builtin_amdgcn_tensor_load_to_lds)
#    define FT_HAS_TDM 1
#  endif
#endif
#ifndef FT_HAS_TDM
#  define FT_HAS_TDM 0
#endif

// -------------------- kernel 1: ragged mean pool, f32 -> bf16 ---------------
__global__ __launch_bounds__(256)
void ft_pool_kernel(const float* __restrict__ x, const float* __restrict__ ts,
                    __bf16* __restrict__ A) {
    const int token = blockIdx.x;           // 0..4095
    const int b = token >> 9;               // token / 512
    const int n = token & 511;

    const float t0 = ts[((size_t)b * Nq + n) * 2 + 0];
    const float t1 = ts[((size_t)b * Nq + n) * 2 + 1];
    int s = (int)floorf(t0 * FRATE);
    int e = (int)floorf(t1 * FRATE);
    e = (e < s + 1) ? (s + 1) : e;
    e = (e > Tq) ? Tq : e;
    const float inv = 1.0f / (float)(e - s);

    const float* xb = x + (size_t)b * Tq * ENC;
    const int d0 = threadIdx.x * 4;         // 256 threads * 4 = 1024 dims

    float4 acc = make_float4(0.f, 0.f, 0.f, 0.f);
    for (int f = s; f < e; ++f) {
        const float4 v = *(const float4*)(xb + (size_t)f * ENC + d0);
        acc.x += v.x; acc.y += v.y; acc.z += v.z; acc.w += v.w;
    }
    __bf16* dst = A + (size_t)token * ENC + d0;
    dst[0] = (__bf16)(acc.x * inv);
    dst[1] = (__bf16)(acc.y * inv);
    dst[2] = (__bf16)(acc.z * inv);
    dst[3] = (__bf16)(acc.w * inv);
}

// -------------------- kernel 2: W f32 -> bf16 -------------------------------
__global__ __launch_bounds__(256)
void ft_convw_kernel(const float* __restrict__ W, __bf16* __restrict__ Wbf) {
    const size_t i0 = ((size_t)blockIdx.x * 256 + threadIdx.x) * 4;
    const float4 v = *(const float4*)(W + i0);
    Wbf[i0 + 0] = (__bf16)v.x;
    Wbf[i0 + 1] = (__bf16)v.y;
    Wbf[i0 + 2] = (__bf16)v.z;
    Wbf[i0 + 3] = (__bf16)v.w;
}

// -------------------- TDM: 2D tile load  global -> LDS ----------------------
// Builds D# per CDNA5 ISA ch.8: group0 {count=1, lds_addr, global_addr, type=2},
// group1 {data_size=2B, tensor dims, tile dims, dim0 stride}. 2D: groups 2/3 zero.
__device__ __forceinline__
void ft_tdm_load_2d(unsigned lds_off, const void* gptr,
                    unsigned tile_k, unsigned tile_rows,
                    unsigned tensor_k, unsigned tensor_rows,
                    unsigned row_stride_elems) {
#if FT_HAS_TDM
    const unsigned long long ga = (unsigned long long)(size_t)gptr;
    v4u g0;
    g0[0] = 1u;                                        // count=1, user descriptor
    g0[1] = lds_off;                                   // lds_addr (bytes)
    g0[2] = (unsigned)(ga & 0xFFFFFFFFu);              // global_addr[31:0]
    g0[3] = (unsigned)((ga >> 32) & 0x01FFFFFFu)       // global_addr[56:32]
          | (2u << 30);                                // type = 2 ("image")
    v8i g1;
    g1[0] = (int)(1u << 16);                           // data_size=1 -> 2 bytes
    g1[1] = (int)((tensor_k & 0xFFFFu) << 16);         // tensor_dim0[15:0]
    g1[2] = (int)((tensor_k >> 16)
          | ((tensor_rows & 0xFFFFu) << 16));          // dim0[31:16] | dim1[15:0]
    g1[3] = (int)((tensor_rows >> 16)
          | ((tile_k & 0xFFFFu) << 16));               // dim1[31:16] | tile_dim0
    g1[4] = (int)(tile_rows & 0xFFFFu);                // tile_dim1 (tile_dim2=0)
    g1[5] = (int)row_stride_elems;                     // tensor_dim0_stride[31:0]
    g1[6] = 0;
    g1[7] = 0;
    v4i g2 = {0, 0, 0, 0};
    v4i g3 = {0, 0, 0, 0};
#  if __clang_major__ >= 23
    v8i g4 = {0, 0, 0, 0, 0, 0, 0, 0};
    __builtin_amdgcn_tensor_load_to_lds(g0, g1, g2, g3, g4, 0);
#  else
    __builtin_amdgcn_tensor_load_to_lds(g0, g1, g2, g3, 0);
#  endif
#else
    (void)lds_off; (void)gptr; (void)tile_k; (void)tile_rows;
    (void)tensor_k; (void)tensor_rows; (void)row_stride_elems;
#endif
}

// -------------------- LDS fragment load (bf16 16x32, K-major, ld=KSTEP) -----
// lanes 0-15 : row = lane,    elems 0-7 -> K=k0+0..7,  elems 8-15 -> K=k0+16..23
// lanes 16-31: row = lane-16, elems 0-7 -> K=k0+8..15, elems 8-15 -> K=k0+24..31
__device__ __forceinline__
v16bf ft_lds_frag(const __bf16* base, int row0, int k0) {
    const int lane = (int)(threadIdx.x & 31);
    const int r  = row0 + (lane & 15);
    const int kk = k0 + ((lane >> 4) << 3);
    const __bf16* p = base + r * KSTEP + kk;
    const v8bf lo = *(const v8bf*)(p);
    const v8bf hi = *(const v8bf*)(p + 16);
    v16bf f;
#pragma unroll
    for (int i = 0; i < 8; ++i) { f[i] = lo[i]; f[i + 8] = hi[i]; }
    return f;
}

// -------------------- kernel 3: bf16 WMMA GEMM (TDM + LDS) + bias + ReLU ----
// C[m,t] = relu( sum_k A[m,k]*W[t,k] + bias[t] ), A:[4096x1024], W:[4096x1024]
// Block = 256 threads = 8 waves (2 x 4). Wave tile 32x32 (2x2 WMMA).
// Block tile 64 x 128, K staged 64 at a time, double-buffered in LDS.
__global__ __launch_bounds__(256)
void ft_gemm_kernel(const __bf16* __restrict__ A, const __bf16* __restrict__ Bm,
                    const float* __restrict__ bias, float* __restrict__ out) {
    __shared__ __bf16 sA[2][BM * KSTEP];   // 2 x 8 KB
    __shared__ __bf16 sB[2][BN * KSTEP];   // 2 x 16 KB

    const int lane  = (int)(threadIdx.x & 31);
    const int wave  = (int)(threadIdx.x >> 5);     // 0..7
    const int waveM = wave & 1;                    // 0..1
    const int waveN = wave >> 1;                   // 0..3

    const int mblk  = (int)blockIdx.x * BM;
    const int nblk  = (int)blockIdx.y * BN;
    const int mbase = waveM * 32;                  // within block tile
    const int nbase = waveN * 32;

    v8f acc00 = {}; v8f acc01 = {}; v8f acc10 = {}; v8f acc11 = {};

#if FT_HAS_TDM
    // producer: wave 0 issues one TDM per tile per K-chunk
    if (wave == 0) {
        ft_tdm_load_2d((unsigned)(size_t)(const void*)&sA[0][0],
                       A + (size_t)mblk * ENC, KSTEP, BM, ENC, MTOT, ENC);
        ft_tdm_load_2d((unsigned)(size_t)(const void*)&sB[0][0],
                       Bm + (size_t)nblk * ENC, KSTEP, BN, ENC, TOK, ENC);
    }
#endif

    for (int kb = 0; kb < NKB; ++kb) {
        const int buf = kb & 1;
#if FT_HAS_TDM
        if (wave == 0) {
            if (kb + 1 < NKB) {
                const int nb = buf ^ 1;
                ft_tdm_load_2d((unsigned)(size_t)(const void*)&sA[nb][0],
                               A + (size_t)mblk * ENC + (size_t)(kb + 1) * KSTEP,
                               KSTEP, BM, ENC, MTOT, ENC);
                ft_tdm_load_2d((unsigned)(size_t)(const void*)&sB[nb][0],
                               Bm + (size_t)nblk * ENC + (size_t)(kb + 1) * KSTEP,
                               KSTEP, BN, ENC, TOK, ENC);
                __builtin_amdgcn_s_wait_tensorcnt(2);   // chunk kb landed
            } else {
                __builtin_amdgcn_s_wait_tensorcnt(0);   // final chunk landed
            }
        }
        __syncthreads();                                // release consumers
#else
        // fallback: cooperative vector copy global -> LDS
        for (int idx = (int)threadIdx.x; idx < BM * KSTEP / 8; idx += 256) {
            const int r = idx / (KSTEP / 8);
            const int c = (idx % (KSTEP / 8)) * 8;
            *(v8bf*)&sA[buf][r * KSTEP + c] =
                *(const v8bf*)(A + (size_t)(mblk + r) * ENC + kb * KSTEP + c);
        }
        for (int idx = (int)threadIdx.x; idx < BN * KSTEP / 8; idx += 256) {
            const int r = idx / (KSTEP / 8);
            const int c = (idx % (KSTEP / 8)) * 8;
            *(v8bf*)&sB[buf][r * KSTEP + c] =
                *(const v8bf*)(Bm + (size_t)(nblk + r) * ENC + kb * KSTEP + c);
        }
        __syncthreads();
#endif
        const __bf16* lA = &sA[buf][0];
        const __bf16* lB = &sB[buf][0];
#pragma unroll
        for (int kk = 0; kk < KSTEP; kk += 32) {
            const v16bf a0 = ft_lds_frag(lA, mbase,      kk);
            const v16bf a1 = ft_lds_frag(lA, mbase + 16, kk);
            const v16bf b0 = ft_lds_frag(lB, nbase,      kk);
            const v16bf b1 = ft_lds_frag(lB, nbase + 16, kk);

            acc00 = __builtin_amdgcn_wmma_f32_16x16x32_bf16(false, a0, false, b0,
                                                            (short)0, acc00, false, false);
            acc01 = __builtin_amdgcn_wmma_f32_16x16x32_bf16(false, a0, false, b1,
                                                            (short)0, acc01, false, false);
            acc10 = __builtin_amdgcn_wmma_f32_16x16x32_bf16(false, a1, false, b0,
                                                            (short)0, acc10, false, false);
            acc11 = __builtin_amdgcn_wmma_f32_16x16x32_bf16(false, a1, false, b1,
                                                            (short)0, acc11, false, false);
        }
        __syncthreads();                                // buffer reusable
    }

    // C/D layout: lane L -> column (L & 15); VGPR v -> row v (+8 if L >= 16)
    const int nloc = lane & 15;
    const int moff = (lane >> 4) << 3;                  // 0 or 8

    const v8f* accs[2][2] = { { &acc00, &acc01 }, { &acc10, &acc11 } };
#pragma unroll
    for (int mi = 0; mi < 2; ++mi) {
#pragma unroll
        for (int ni = 0; ni < 2; ++ni) {
            const int   nn = nblk + nbase + ni * 16 + nloc;
            const float bv = bias[nn];
            const v8f&  a  = *accs[mi][ni];
#pragma unroll
            for (int v = 0; v < 8; ++v) {
                const int mm = mblk + mbase + mi * 16 + moff + v;
                float h = a[v] + bv;
                h = (h > 0.f) ? h : 0.f;
                out[(size_t)mm * TOK + nn] = h;
            }
        }
    }
}

// -------------------- kernel 4: in-place LayerNorm over TOK ----------------
__global__ __launch_bounds__(256)
void ft_ln_kernel(float* __restrict__ out, const float* __restrict__ gamma,
                  const float* __restrict__ beta) {
    const int row = blockIdx.x;
    float* h = out + (size_t)row * TOK;
    const int tid = (int)threadIdx.x;

    float vals[16];
    float lsum = 0.f, lsq = 0.f;
#pragma unroll
    for (int i = 0; i < 16; ++i) {
        const float v = h[tid + i * 256];
        vals[i] = v;
        lsum += v;
        lsq  += v * v;
    }

    __shared__ float ssum[256];
    __shared__ float ssq[256];
    ssum[tid] = lsum; ssq[tid] = lsq;
    __syncthreads();
#pragma unroll
    for (int off = 128; off > 0; off >>= 1) {
        if (tid < off) { ssum[tid] += ssum[tid + off]; ssq[tid] += ssq[tid + off]; }
        __syncthreads();
    }
    const float mu  = ssum[0] * (1.0f / TOK);
    const float var = ssq[0] * (1.0f / TOK) - mu * mu;
    const float rs  = rsqrtf(var + LN_EPS);

#pragma unroll
    for (int i = 0; i < 16; ++i) {
        const int t = tid + i * 256;
        h[t] = (vals[i] - mu) * rs * gamma[t] + beta[t];
    }
}

// ---------------------------------------------------------------------------
extern "C" void kernel_launch(void* const* d_in, const int* in_sizes, int n_in,
                              void* d_out, int out_size, void* d_ws, size_t ws_size,
                              hipStream_t stream) {
    (void)in_sizes; (void)n_in; (void)out_size; (void)ws_size;

    const float* x     = (const float*)d_in[0];   // [8,3000,1024]
    const float* ts    = (const float*)d_in[1];   // [8,512,2]
    const float* W     = (const float*)d_in[2];   // [4096,1024]
    const float* bias  = (const float*)d_in[3];   // [4096]
    const float* gamma = (const float*)d_in[4];   // [4096]
    const float* beta  = (const float*)d_in[5];   // [4096]
    float*       out   = (float*)d_out;           // [8,512,4096]

    // workspace: A_bf16 [4096x1024] @ +0 (8 MB), W_bf16 [4096x1024] @ +8 MB
    __bf16* Abf = (__bf16*)d_ws;
    __bf16* Wbf = (__bf16*)((char*)d_ws + (size_t)8 * 1024 * 1024);

    ft_pool_kernel<<<MTOT, 256, 0, stream>>>(x, ts, Abf);
    ft_convw_kernel<<<(TOK * ENC) / (256 * 4), 256, 0, stream>>>(W, Wbf);

    dim3 ggrid(MTOT / BM, TOK / BN);
    ft_gemm_kernel<<<ggrid, 256, 0, stream>>>(Abf, Wbf, bias, out);

    ft_ln_kernel<<<MTOT, 256, 0, stream>>>(out, gamma, beta);
}
